// QuantizedBlock_79508434583566
// MI455X (gfx1250) — matrix operand order repaired and checked by reading, not compiled, not run on hardware
//
#include <hip/hip_runtime.h>
#include <hip/hip_bf16.h>
#include <math.h>

typedef __attribute__((ext_vector_type(16))) _Float16 v16h;
typedef __attribute__((ext_vector_type(8)))  _Float16 v8h;
typedef __attribute__((ext_vector_type(8)))  float    v8f;

#define B_N   2048
#define K_DIM 2048
#define N_DIM 2048
#define NC    4
#define EPSQ  1e-9f
#define BNEPS 1e-5f

#define TM 128
#define TN 128
#define TK 32
#define LDSTR 40   // halves per LDS row (32 data + 8 pad, keeps 16B alignment)

// ---------------- init workspace ----------------
__global__ void init_ws_kernel(float* absmax) {
    if (threadIdx.x == 0 && blockIdx.x == 0) absmax[0] = 0.0f;
}

// ---------------- max|W| reduction ----------------
__global__ void absmax_kernel(const float* __restrict__ W, float* __restrict__ absmax) {
    __shared__ float red[256];
    const int tid = threadIdx.x;
    float m = 0.0f;
    for (long i = (long)blockIdx.x * 256 + tid; i < (long)N_DIM * K_DIM; i += (long)gridDim.x * 256)
        m = fmaxf(m, fabsf(W[i]));
    red[tid] = m;
    __syncthreads();
    for (int s = 128; s > 0; s >>= 1) {
        if (tid < s) red[tid] = fmaxf(red[tid], red[tid + s]);
        __syncthreads();
    }
    // all values >= 0: uint ordering == float ordering -> exact, deterministic
    if (tid == 0) atomicMax((unsigned int*)absmax, __float_as_uint(red[0]));
}

// ---------------- per-sample controller ----------------
__global__ void controller_kernel(const float* __restrict__ x,
                                  const float* __restrict__ gu,
                                  const float* __restrict__ temp,
                                  const float* __restrict__ Wc1, const float* __restrict__ bc1,
                                  const float* __restrict__ Wc2, const float* __restrict__ bc2,
                                  int* __restrict__ choice,
                                  float* __restrict__ logp_out,
                                  float* __restrict__ bits_out) {
    __shared__ float ssum[256], ssq[256], szero[256];
    const int b = blockIdx.x, tid = threadIdx.x;
    const float* row = x + (size_t)b * K_DIM;
    float s = 0.f, sq = 0.f, z = 0.f;
    for (int k = tid; k < K_DIM; k += 256) {
        float v = row[k];
        s += v; sq += v * v; z += (v == 0.0f) ? 1.0f : 0.0f;
    }
    ssum[tid] = s; ssq[tid] = sq; szero[tid] = z;
    __syncthreads();
    for (int st = 128; st > 0; st >>= 1) {
        if (tid < st) { ssum[tid] += ssum[tid+st]; ssq[tid] += ssq[tid+st]; szero[tid] += szero[tid+st]; }
        __syncthreads();
    }
    if (tid == 0) {
        const float mean = ssum[0] / (float)K_DIM;
        const float var  = (ssq[0] - (float)K_DIM * mean * mean) / (float)(K_DIM - 1); // ddof=1
        const float zf   = szero[0] / (float)K_DIM;
        float h[16];
        #pragma unroll
        for (int j = 0; j < 16; j++) {
            float a = Wc1[j*3+0]*mean + Wc1[j*3+1]*var + Wc1[j*3+2]*zf + bc1[j];
            h[j] = a > 0.f ? a : 0.f;
        }
        float logits[NC];
        #pragma unroll
        for (int c = 0; c < NC; c++) {
            float a = bc2[c];
            #pragma unroll
            for (int j = 0; j < 16; j++) a += Wc2[c*16+j] * h[j];
            logits[c] = a;
        }
        const float t = temp[0];
        int best = 0; float bv = (logits[0] + gu[b*NC+0]) / t;
        #pragma unroll
        for (int c = 1; c < NC; c++) {
            float v = (logits[c] + gu[b*NC+c]) / t;
            if (v > bv) { bv = v; best = c; }
        }
        float m = logits[0];
        #pragma unroll
        for (int c = 1; c < NC; c++) m = fmaxf(m, logits[c]);
        float den = 0.f;
        #pragma unroll
        for (int c = 0; c < NC; c++) den += expf(logits[c] - m);
        const float p = expf(logits[best] - m) / den;
        choice[b]   = best;
        logp_out[b] = logf(p + EPSQ);
        const int bitsArr[NC] = {2, 4, 8, 32};
        bits_out[b] = (float)bitsArr[best];
    }
}

// ---------------- group rows by choice (padded with -1) ----------------
__global__ void build_perm_kernel(const int* __restrict__ choice, int* __restrict__ perm) {
    __shared__ int cnt[NC];
    const int tid = threadIdx.x;
    if (tid < NC) cnt[tid] = 0;
    for (int i = tid; i < NC * B_N; i += 256) perm[i] = -1;
    __syncthreads();
    for (int i = tid; i < B_N; i += 256) {
        const int c = choice[i];
        const int pos = atomicAdd(&cnt[c], 1);
        perm[c * B_N + pos] = i;
    }
}

// ---------------- grouped fake-quant WMMA GEMM: out[r] = x[r] @ Wq(seg)^T + bias ----------------
// 256 threads = 8 waves; block tile 128x128; wave tile 32x64 = 2x4 WMMA fragments.
// Per K-step per wave: 12 ds_load_b128 feed 8 v_wmma (1.5 LDS loads / WMMA).
__global__ __launch_bounds__(256)
void gemm_kernel(const float* __restrict__ x, const float* __restrict__ W,
                 const float* __restrict__ bias, const float* __restrict__ absmax,
                 const int* __restrict__ perm, float* __restrict__ out) {
    __shared__ __align__(16) _Float16 As[TM * LDSTR];
    __shared__ __align__(16) _Float16 Bs[TN * LDSTR];
    __shared__ int sPerm[TM];

    const int seg   = blockIdx.y >> 4;   // 4 segments x 16 M-tiles
    const int mtile = blockIdx.y & 15;
    const int n0    = blockIdx.x * TN;
    const int* segPerm = perm + seg * B_N + mtile * TM;
    if (segPerm[0] < 0) return;          // fully padded tile (uniform exit)

    const int tid  = threadIdx.x;
    if (tid < TM) sPerm[tid] = segPerm[tid];

    // per-segment quantization parameters (uniform within block)
    float scale = 1.0f, inv = 0.0f;
    const bool doq = (seg < 3);
    if (doq) {
        const int bitsArr[3] = {2, 4, 8};
        const float qlev = (float)((1u << bitsArr[seg]) - 1u);
        const float amax = absmax[0];
        scale = 2.0f * amax / (qlev + EPSQ);
        inv   = 1.0f / (scale + EPSQ);
    }

    const int lane = tid & 31;
    const int wv   = tid >> 5;
    const int waveM = (wv >> 1) * 32;    // 0,32,64,96
    const int waveN = (wv & 1) * 64;     // 0,64

    v8f acc[2][4];
    #pragma unroll
    for (int i = 0; i < 2; i++)
        #pragma unroll
        for (int j = 0; j < 4; j++)
            acc[i][j] = (v8f){0.f,0.f,0.f,0.f,0.f,0.f,0.f,0.f};

    __syncthreads();  // sPerm visible

    // cooperative-load assignments: thread t fills row t/2, half (t&1)*16 of the K-tile
    const int lrow = tid >> 1;           // 0..127
    const int lcb  = (tid & 1) * 16;     // 0 or 16
    int gxrow = sPerm[lrow];
    if (gxrow < 0) gxrow = 0;            // padded row: load anything, store skipped
    const float* xrow = x + (size_t)gxrow * K_DIM;
    const float* wrow = W + (size_t)(n0 + lrow) * K_DIM;

    // fragment addressing per documented 16-bit WMMA layouts
    const int am  = lane & 15;
    const int kbA = (lane < 16) ? 0 : 8;   // A: halves 0-7 -> K kbA..kbA+7, halves 8-15 -> K kbA+16..
    const int bn  = lane & 15;
    const int kbB = (lane < 16) ? 0 : 8;   // B: halves 0-7 -> K kbB..kbB+7, halves 8-15 -> K kbB+16..

    for (int k0 = 0; k0 < K_DIM; k0 += TK) {
        // stage A (x rows -> f16) and B (quantized W rows -> f16) into LDS
        __align__(16) _Float16 ta[16];
        __align__(16) _Float16 tb[16];
        __builtin_prefetch(xrow + k0 + TK, 0, 0);
        __builtin_prefetch(wrow + k0 + TK, 0, 0);
        #pragma unroll
        for (int i = 0; i < 16; i++) ta[i] = (_Float16)xrow[k0 + lcb + i];
        #pragma unroll
        for (int i = 0; i < 16; i++) {
            float w = wrow[k0 + lcb + i];
            if (doq) w = rintf(w * inv) * scale;   // fake_quant: round(w/(s+eps))*s
            tb[i] = (_Float16)w;
        }
        *(v8h*)&As[lrow * LDSTR + lcb]     = *(v8h*)&ta[0];
        *(v8h*)&As[lrow * LDSTR + lcb + 8] = *(v8h*)&ta[8];
        *(v8h*)&Bs[lrow * LDSTR + lcb]     = *(v8h*)&tb[0];
        *(v8h*)&Bs[lrow * LDSTR + lcb + 8] = *(v8h*)&tb[8];
        __syncthreads();

        v16h af[2], bf[4];
        #pragma unroll
        for (int mi = 0; mi < 2; mi++) {
            const v8h alo = *(const v8h*)&As[(waveM + mi*16 + am) * LDSTR + kbA];
            const v8h ahi = *(const v8h*)&As[(waveM + mi*16 + am) * LDSTR + kbA + 16];
            #pragma unroll
            for (int i = 0; i < 8; i++) { af[mi][i] = alo[i]; af[mi][8+i] = ahi[i]; }
        }
        #pragma unroll
        for (int ni = 0; ni < 4; ni++) {
            const v8h blo = *(const v8h*)&Bs[(waveN + ni*16 + bn) * LDSTR + kbB];
            const v8h bhi = *(const v8h*)&Bs[(waveN + ni*16 + bn) * LDSTR + kbB + 16];
            #pragma unroll
            for (int i = 0; i < 8; i++) { bf[ni][i] = blo[i]; bf[ni][8+i] = bhi[i]; }
        }
        #pragma unroll
        for (int mi = 0; mi < 2; mi++)
            #pragma unroll
            for (int ni = 0; ni < 4; ni++)
                acc[mi][ni] = __builtin_amdgcn_wmma_f32_16x16x32_f16(
                    false, af[mi], false, bf[ni], (short)0, acc[mi][ni], false, false);
        __syncthreads();
    }

    // epilogue: C/D layout -> lane<16: M=v, lane>=16: M=v+8; N = lane%16
    const int cn = lane & 15;
    const int moff = (lane < 16) ? 0 : 8;
    #pragma unroll
    for (int mi = 0; mi < 2; mi++) {
        #pragma unroll
        for (int ni = 0; ni < 4; ni++) {
            #pragma unroll
            for (int v = 0; v < 8; v++) {
                const int mlocal = waveM + mi*16 + moff + v;
                const int gr = sPerm[mlocal];
                if (gr >= 0) {
                    const int gn = n0 + waveN + ni*16 + cn;
                    out[(size_t)gr * N_DIM + gn] = acc[mi][ni][v] + bias[gn];
                }
            }
        }
    }
}

// ---------------- deterministic per-column stats (no float atomics) ----------------
__global__ void colstats_kernel(const float* __restrict__ y,
                                float* __restrict__ colsum, float* __restrict__ colsumsq) {
    const int col = blockIdx.x * 256 + threadIdx.x;
    float s = 0.f, sq = 0.f;
    for (int r = 0; r < B_N; r++) {
        const float v = y[(size_t)r * N_DIM + col];
        s += v; sq += v * v;
    }
    colsum[col] = s;
    colsumsq[col] = sq;
}

// ---------------- in-place BatchNorm(train, biased var) + ReLU ----------------
__global__ void bn_relu_kernel(float* __restrict__ y,
                               const float* __restrict__ colsum, const float* __restrict__ colsumsq,
                               const float* __restrict__ gamma, const float* __restrict__ beta) {
    const size_t i = (size_t)blockIdx.x * 256 + threadIdx.x;
    const int col = (int)(i & (N_DIM - 1));
    const float mean = colsum[col] / (float)B_N;
    const float var  = colsumsq[col] / (float)B_N - mean * mean;
    const float rstd = rsqrtf(var + BNEPS);
    const float v = (y[i] - mean) * rstd * gamma[col] + beta[col];
    y[i] = v > 0.f ? v : 0.f;
}

extern "C" void kernel_launch(void* const* d_in, const int* in_sizes, int n_in,
                              void* d_out, int out_size, void* d_ws, size_t ws_size,
                              hipStream_t stream) {
    const float* x    = (const float*)d_in[0];
    const float* gu   = (const float*)d_in[1];
    const float* temp = (const float*)d_in[2];
    const float* W    = (const float*)d_in[3];
    const float* bias = (const float*)d_in[4];
    const float* gamma= (const float*)d_in[5];
    const float* beta = (const float*)d_in[6];
    const float* Wc1  = (const float*)d_in[7];
    const float* bc1  = (const float*)d_in[8];
    const float* Wc2  = (const float*)d_in[9];
    const float* bc2  = (const float*)d_in[10];

    float* y       = (float*)d_out;
    float* logp    = y + (size_t)B_N * N_DIM;
    float* bitsOut = logp + B_N;

    char* ws = (char*)d_ws;
    float* absmax   = (float*)ws;                        // 16 B slot
    int*   choice   = (int*)(ws + 16);                   // 8 KB
    int*   perm     = (int*)(ws + 16 + 8192);            // 32 KB
    float* colsum   = (float*)(ws + 16 + 8192 + 32768);  // 8 KB
    float* colsumsq = colsum + N_DIM;                    // 8 KB

    init_ws_kernel<<<1, 1, 0, stream>>>(absmax);
    absmax_kernel<<<512, 256, 0, stream>>>(W, absmax);
    controller_kernel<<<B_N, 256, 0, stream>>>(x, gu, temp, Wc1, bc1, Wc2, bc2,
                                               choice, logp, bitsOut);
    build_perm_kernel<<<1, 256, 0, stream>>>(choice, perm);
    gemm_kernel<<<dim3(N_DIM / TN, NC * (B_N / TM)), 256, 0, stream>>>(
        x, W, bias, absmax, perm, y);
    colstats_kernel<<<N_DIM / 256, 256, 0, stream>>>(y, colsum, colsumsq);
    bn_relu_kernel<<<(int)(((size_t)B_N * N_DIM) / 256), 256, 0, stream>>>(
        y, colsum, colsumsq, gamma, beta);
}
